// GCN_decay_rep2_35871566856582
// MI455X (gfx1250) — matrix operand order repaired and checked by reading, not compiled, not run on hardware
//
#include <hip/hip_runtime.h>
#include <hip/hip_bf16.h>

// ---------------------------------------------------------------------------
// CDNA5 (gfx1250) GCN forward.
//  - Dense layers: bf16 WMMA (v_wmma_f32_16x16x32_bf16), f32 accumulate.
//    W pre-packed to bf16 in WMMA B-fragment order (flat: one 32B load per
//    fragment, next tile always +1KB -> software-pipelined lookahead).
//    A fragments loaded straight from global (fragment layout partitions
//    rows/k disjointly across lanes), double-buffered across k-steps,
//    converted with __builtin_convertvector (hw bf16 cvt). No LDS/barriers.
//  - SpMM: 1 wave per edge, float4 gather + global_atomic_add_f32 scatter.
//  - ReLU fused into consumer GEMM's A load; bias fused into SpMM init /
//    GEMM epilogue; concats realized as K-chunked accumulating GEMM calls.
// Workspace: 7 * N*128 f32 buffers (~179 MB) + 97 KB packed-W scratch.
// ---------------------------------------------------------------------------

typedef __attribute__((ext_vector_type(16))) __bf16 v16bf;
typedef __attribute__((ext_vector_type(16))) float  v16f;
typedef __attribute__((ext_vector_type(8)))  float  v8f;

// ---------------------------------------------------------------------------
// Pack W[K x M] (f32, row stride Wstride) into bf16 WMMA B-fragments:
//   out[(((ks*MT + t)*32 + lane)*16 + p)] = bf16( W[ks*32 + half*8 + p + (p>=8 ? 8 : 0)][t*16+lane16] )
// ---------------------------------------------------------------------------
__global__ __launch_bounds__(256) void pack_w(
    const float* __restrict__ W, int Wstride,
    unsigned short* __restrict__ out, int K, int M)
{
  const long long total = (long long)K * M;
  long long i = (long long)blockIdx.x * 256 + threadIdx.x;
  if (i >= total) return;
  const int MT = M >> 4;
  int p    = (int)(i & 15);
  int lane = (int)((i >> 4) & 31);
  long long g = i >> 9;                 // (ks*MT + t)
  int t  = (int)(g % MT);
  int ks = (int)(g / MT);
  int half = lane >> 4, lane16 = lane & 15;
  int krow = ks * 32 + half * 8 + p + (p >= 8 ? 8 : 0);
  int col  = t * 16 + lane16;
  union { __bf16 b; unsigned short s; } cv;
  cv.b = (__bf16)W[(size_t)krow * Wstride + col];
  out[i] = cv.s;
}

// ---------------------------------------------------------------------------
// GEMM: out[N x M] (+)= act(A[N x K]) @ W[K x M] (+ bias)
//  - 256 threads = 8 waves; each wave owns 2 strips of 16 rows
//    (block tile = 256 rows x M); B fragments reused across both strips.
//  - A rows clamped (no exec divergence in the hot loop); stores guarded.
//  - A raw loads double-buffered across k-steps; B fragments pipelined one
//    tile ahead (flat packed layout makes "next" always +32 v16bf).
// ---------------------------------------------------------------------------
template<int M, bool RELUIN, bool BIAS, bool ACCUM>
__global__ __launch_bounds__(256) void gemm_wmma(
    const float* __restrict__ A, int Astride,
    const v16bf* __restrict__ Wfrag,
    const float* __restrict__ bias,
    float* __restrict__ out, int outStride,
    int N, int K)
{
  constexpr int MT = M / 16;
  constexpr int RS = 2;                       // row strips per wave

  const int tid    = threadIdx.x;
  const int w      = tid >> 5;
  const int lane   = tid & 31;
  const int half   = lane >> 4;
  const int lane16 = lane & 15;
  const int row0   = blockIdx.x * (128 * RS) + w * (16 * RS);

  v8f acc[RS][MT];
#pragma unroll
  for (int s = 0; s < RS; ++s)
#pragma unroll
    for (int t = 0; t < MT; ++t)
      acc[s][t] = (v8f){0.f,0.f,0.f,0.f,0.f,0.f,0.f,0.f};

  // clamp A row: garbage rows only feed C rows >= N, which are never stored
  const float* aptr[RS];
#pragma unroll
  for (int s = 0; s < RS; ++s) {
    int ar = row0 + s * 16 + lane16;
    if (ar > N - 1) ar = N - 1;
    aptr[s] = A + (size_t)ar * Astride + half * 8;
  }

  const int nks = K >> 5;

  union AU { v16f v; float f[16]; float4 q[4]; };
  AU araw[RS];
  // preload A raw for ks=0:  positions 0-7: k=half*8+0..7 ; 8-15: +16
#pragma unroll
  for (int s = 0; s < RS; ++s) {
    araw[s].q[0] = ((const float4*)(aptr[s]))[0];
    araw[s].q[1] = ((const float4*)(aptr[s]))[1];
    araw[s].q[2] = ((const float4*)(aptr[s] + 16))[0];
    araw[s].q[3] = ((const float4*)(aptr[s] + 16))[1];
  }
  // preload first B fragment
  const v16bf* wpl = Wfrag + lane;
  v16bf bcur = wpl[0];

  for (int ks = 0; ks < nks; ++ks) {
    // convert current A raw -> bf16 fragments
    v16bf af[RS];
#pragma unroll
    for (int s = 0; s < RS; ++s) {
      AU u = araw[s];
      if (RELUIN) {
#pragma unroll
        for (int i = 0; i < 16; ++i) u.f[i] = fmaxf(u.f[i], 0.f);
      }
      af[s] = __builtin_convertvector(u.v, v16bf);
    }
    // prefetch A raw for next k-step (clamped to last step)
    {
      const int kn = (ks + 1 < nks) ? (ks + 1) << 5 : ks << 5;
#pragma unroll
      for (int s = 0; s < RS; ++s) {
        const float* ap = aptr[s] + kn;
        araw[s].q[0] = ((const float4*)ap)[0];
        araw[s].q[1] = ((const float4*)ap)[1];
        araw[s].q[2] = ((const float4*)(ap + 16))[0];
        araw[s].q[3] = ((const float4*)(ap + 16))[1];
      }
    }
    // B one-tile lookahead + WMMA (flat layout: next tile = +32 v16bf;
    // pack buffer is over-allocated by one tile so the last read is safe)
    const v16bf* wpt = wpl + (size_t)ks * MT * 32;
#pragma unroll
    for (int t = 0; t < MT; ++t) {
      v16bf bnext = wpt[(t + 1) * 32];
#pragma unroll
      for (int s = 0; s < RS; ++s)
        acc[s][t] = __builtin_amdgcn_wmma_f32_16x16x32_bf16(
            false, af[s], false, bcur, (short)0, acc[s][t], false, false);
      bcur = bnext;
    }
  }

  // ---- epilogue: C layout -> row = strip + 8*half + r, col = 16t + lane16 ----
#pragma unroll
  for (int s = 0; s < RS; ++s) {
#pragma unroll
    for (int t = 0; t < MT; ++t) {
      const int col = t * 16 + lane16;
#pragma unroll
      for (int r = 0; r < 8; ++r) {
        const int row = row0 + s * 16 + half * 8 + r;
        if (row < N) {
          float v = acc[s][t][r];
          if (BIAS) v += bias[col];
          size_t o = (size_t)row * outStride + col;
          if (ACCUM) v += out[o];
          out[o] = v;
        }
      }
    }
  }
}

// ---------------------------------------------------------------------------
// SpMM: out[dst] += val * h[src]  (128 features; 1 wave per edge, float4/lane)
// ---------------------------------------------------------------------------
__global__ __launch_bounds__(256) void spmm_edges(
    const int* __restrict__ edges, const float* __restrict__ vals,
    const float* __restrict__ h, float* __restrict__ out, int E)
{
  const int gt   = blockIdx.x * 256 + threadIdx.x;
  const int e    = gt >> 5;
  const int lane = threadIdx.x & 31;
  if (e >= E) return;
  const int   src = edges[e];
  const int   dst = edges[(size_t)E + e];
  const float v   = vals[e];
  const float4 q  = ((const float4*)(h + (size_t)src * 128))[lane];
  float* o = out + (size_t)dst * 128 + lane * 4;
  atomicAdd(o + 0, v * q.x);
  atomicAdd(o + 1, v * q.y);
  atomicAdd(o + 2, v * q.z);
  atomicAdd(o + 3, v * q.w);
}

// out[i][j] = bias[j]  for N x 128
__global__ __launch_bounds__(256) void fill_bias128(
    float* __restrict__ out, const float* __restrict__ bias, long long total)
{
  long long i = (long long)blockIdx.x * 256 + threadIdx.x;
  if (i < total) out[i] = bias[i & 127];
}

// ---------------------------------------------------------------------------
// host-side dispatch
// ---------------------------------------------------------------------------
template<int M>
static void gemm_m(hipStream_t s, const float* A, int Asr, const v16bf* Wf,
                   const float* bias, float* out, int osr, int N, int K,
                   bool reluIn, bool accum)
{
  dim3 g((N + 255) / 256), b(256);
  int sel = (reluIn ? 4 : 0) | (bias ? 2 : 0) | (accum ? 1 : 0);
  switch (sel) {
    case 0: gemm_wmma<M,false,false,false><<<g,b,0,s>>>(A,Asr,Wf,bias,out,osr,N,K); break;
    case 1: gemm_wmma<M,false,false,true ><<<g,b,0,s>>>(A,Asr,Wf,bias,out,osr,N,K); break;
    case 2: gemm_wmma<M,false,true ,false><<<g,b,0,s>>>(A,Asr,Wf,bias,out,osr,N,K); break;
    case 3: gemm_wmma<M,false,true ,true ><<<g,b,0,s>>>(A,Asr,Wf,bias,out,osr,N,K); break;
    case 4: gemm_wmma<M,true ,false,false><<<g,b,0,s>>>(A,Asr,Wf,bias,out,osr,N,K); break;
    case 5: gemm_wmma<M,true ,false,true ><<<g,b,0,s>>>(A,Asr,Wf,bias,out,osr,N,K); break;
    case 6: gemm_wmma<M,true ,true ,false><<<g,b,0,s>>>(A,Asr,Wf,bias,out,osr,N,K); break;
    default:gemm_wmma<M,true ,true ,true ><<<g,b,0,s>>>(A,Asr,Wf,bias,out,osr,N,K); break;
  }
}

static void gemm(hipStream_t s, const float* A, int Asr,
                 const float* W, int Wsr, unsigned short* wpack,
                 const float* bias, float* out, int osr,
                 int N, int K, int M, bool reluIn, bool accum)
{
  const long long tot = (long long)K * M;
  pack_w<<<dim3((unsigned)((tot + 255) / 256)), dim3(256), 0, s>>>(W, Wsr, wpack, K, M);
  if (M == 128) gemm_m<128>(s, A, Asr, (const v16bf*)wpack, bias, out, osr, N, K, reluIn, accum);
  else          gemm_m<32 >(s, A, Asr, (const v16bf*)wpack, bias, out, osr, N, K, reluIn, accum);
}

extern "C" void kernel_launch(void* const* d_in, const int* in_sizes, int n_in,
                              void* d_out, int out_size, void* d_ws, size_t ws_size,
                              hipStream_t stream)
{
  (void)n_in; (void)out_size; (void)ws_size;
  const float* x = (const float*)d_in[0];
  const int N = in_sizes[0] / 256;
  const int E = in_sizes[1] / 2;

  const int*   edg[6]; const float* val[6];
  for (int i = 0; i < 5; ++i) {
    edg[i + 1] = (const int*)  d_in[1 + 2 * i];
    val[i + 1] = (const float*)d_in[2 + 2 * i];
  }
  // params start at index 11; order: gc1..gc12, sim_dense, ensemble1..3,
  // dense1, dense2, dense3, densesim, dense4   (W then b each)
  auto Wp = [&](int i) { return (const float*)d_in[11 + 2 * i]; };
  auto Bp = [&](int i) { return (const float*)d_in[11 + 2 * i + 1]; };
  enum { GC1=0, GC2, GC3, GC4, GC5, GC6, GC7, GC8, GC9, GC10, GC11, GC12,
         SIMD, ENS1, ENS2, ENS3, D1, D2, D3, DSIM, D4 };

  const size_t bufE = (size_t)N * 128;
  float* B[7];
  for (int i = 0; i < 7; ++i) B[i] = (float*)d_ws + (size_t)i * bufE;
  float *X1 = B[0], *X2 = B[1], *X3 = B[2], *X4 = B[3];
  float *T  = B[4], *HA = B[5], *HB = B[6];
  // packed-W scratch after the big buffers (one extra 1KB tile of slack is
  // required for the B-fragment lookahead; ws has it)
  unsigned short* WPK = (unsigned short*)((float*)d_ws + 7 * bufE);

  float* out = (float*)d_out;
  const size_t ob = (size_t)N * 32;

  const long long totalH = (long long)N * 128;
  dim3 gFill((unsigned)((totalH + 255) / 256)), b256(256);
  dim3 gSp((unsigned)(((size_t)E * 32 + 255) / 256));

  auto spmm_into = [&](const int* e, const float* v, const float* hin,
                       const float* bias, float* o) {
    fill_bias128<<<gFill, b256, 0, stream>>>(o, bias, totalH);
    spmm_edges<<<gSp, b256, 0, stream>>>(e, v, hin, o, E);
  };

  // one GCN branch: X = gc3(relu(gc2(relu(gc1(x)))))   (bias via spmm init,
  // relu fused into the next gemm's A load)
  auto branch = [&](int gA, int gB, int gC, const int* e, const float* v, float* X) {
    gemm(stream, x, 256, Wp(gA), 128, WPK, nullptr, T, 128, N, 256, 128, false, false);
    spmm_into(e, v, T, Bp(gA), HA);
    gemm(stream, HA, 128, Wp(gB), 128, WPK, nullptr, T, 128, N, 128, 128, true, false);
    spmm_into(e, v, T, Bp(gB), HB);
    gemm(stream, HB, 128, Wp(gC), 128, WPK, nullptr, T, 128, N, 128, 128, true, false);
    spmm_into(e, v, T, Bp(gC), X);
  };

  branch(GC1,  GC2,  GC3,  edg[5], val[5], X1);
  branch(GC4,  GC5,  GC6,  edg[4], val[4], X2);
  branch(GC7,  GC8,  GC9,  edg[3], val[3], X3);
  branch(GC10, GC11, GC12, edg[1], val[1], X4);

  // heads (input-relu fused)
  gemm(stream, X1, 128, Wp(D1), 32, WPK, Bp(D1), out + 0 * ob, 32, N, 128, 32, true, false);
  gemm(stream, X2, 128, Wp(D2), 32, WPK, Bp(D2), out + 1 * ob, 32, N, 128, 32, true, false);
  gemm(stream, X3, 128, Wp(D3), 32, WPK, Bp(D3), out + 2 * ob, 32, N, 128, 32, true, false);
  gemm(stream, X4, 128, Wp(D3), 32, WPK, Bp(D3), out + 3 * ob, 32, N, 128, 32, true, false); // reuses dense3

  // xsim = concat(x2,x3) @ simW + b   (K-chunked accumulate; HA free -> XSIM)
  float* XSIM = HA;
  gemm(stream, X2, 128, Wp(SIMD),           128, WPK, Bp(SIMD), XSIM, 128, N, 128, 128, false, false);
  gemm(stream, X3, 128, Wp(SIMD) + 128*128, 128, WPK, nullptr,  XSIM, 128, N, 128, 128, false, true);
  gemm(stream, XSIM, 128, Wp(DSIM), 32, WPK, Bp(DSIM), out + 4 * ob, 32, N, 128, 32, true, false);

  // ensemble chain (X2/X3 buffers now free; relu deferred to consumer A-load)
  float* E1 = X2; float* E2 = X3; float* E3 = T;
  gemm(stream, X1,   128, Wp(ENS1),           128, WPK, Bp(ENS1), E1, 128, N, 128, 128, false, false);
  gemm(stream, XSIM, 128, Wp(ENS1) + 128*128, 128, WPK, nullptr,  E1, 128, N, 128, 128, false, true);
  gemm(stream, X4,   128, Wp(ENS1) + 256*128, 128, WPK, nullptr,  E1, 128, N, 128, 128, false, true);
  gemm(stream, E1, 128, Wp(ENS2), 128, WPK, Bp(ENS2), E2, 128, N, 128, 128, true, false);
  gemm(stream, E2, 128, Wp(ENS3), 128, WPK, Bp(ENS3), E3, 128, N, 128, 128, true, false);
  gemm(stream, E3, 128, Wp(D4),   32,  WPK, Bp(D4), out + 5 * ob, 32, N, 128, 32, true, false);
}
